// Low_Rank_linear_42846593745144
// MI455X (gfx1250) — matrix operand order repaired and checked by reading, not compiled, not run on hardware
//
#include <hip/hip_runtime.h>
#include <hip/hip_bf16.h>

typedef __attribute__((ext_vector_type(16))) __bf16 v16bf;
typedef __attribute__((ext_vector_type(8)))  __bf16 v8bf;
typedef __attribute__((ext_vector_type(8)))  float  v8f;
typedef __attribute__((ext_vector_type(4)))  int    v4i;

#define NFULL 4096
#define RANK  512
#define NCK   3840   // columns kept
#define NCC   256    // column complement
#define K2    768    // RANK + NCC (fused [A | sparse1] K dim)
#define MTOT  8192   // BATCH * SEQ

#define BM 128
#define BN 128
#define KS 32
#define LROW 40      // LDS row pitch in halfs: 32 data + 8 pad (80B, conflict-free b128 frags)

// ---- gfx1250 async global->LDS copy (ASYNCcnt path), guarded -----------------

#if defined(__HIP_DEVICE_COMPILE__) && __has_builtin(__builtin_amdgcn_global_load_async_to_lds_b128)
#define HAVE_ASYNC_LDS 1
#else
#define HAVE_ASYNC_LDS 0
#endif

__device__ __forceinline__ void async_b128(const void* g, void* l) {
#if HAVE_ASYNC_LDS
  __builtin_amdgcn_global_load_async_to_lds_b128(
      (__attribute__((address_space(1))) v4i*)(v4i*)(void*)g,
      (__attribute__((address_space(3))) v4i*)(v4i*)l,
      0, 0);
#else
  (void)g; (void)l;
#endif
}

__device__ __forceinline__ void wait_async0() {
#if defined(__HIP_DEVICE_COMPILE__)
#if __has_builtin(__builtin_amdgcn_s_wait_asynccnt)
  __builtin_amdgcn_s_wait_asynccnt(0);
#else
  asm volatile("s_wait_asynccnt 0x0" ::: "memory");
#endif
#endif
}

__device__ __forceinline__ unsigned short f2bf_bits(float f) {
  __hip_bfloat16 h = __float2bfloat16(f);
  return *reinterpret_cast<unsigned short*>(&h);
}

// ---------------- prep kernels: fp32 -> bf16 weight / activation staging ----------------

__global__ void cvt_f32_bf16(const float* __restrict__ in, __hip_bfloat16* __restrict__ out, int n) {
  int i = blockIdx.x * blockDim.x + threadIdx.x;
  int stride = gridDim.x * blockDim.x;
  for (; i < n; i += stride) out[i] = __float2bfloat16(in[i]);
}

// W2[o, 0:512) = A[o,:], W2[o, 512:768) = sparse1[o,:]
__global__ void build_w2(const float* __restrict__ A, const float* __restrict__ S1,
                         __hip_bfloat16* __restrict__ W2) {
  int i = blockIdx.x * blockDim.x + threadIdx.x;
  int stride = gridDim.x * blockDim.x;
  const int total = NCK * K2;
  for (; i < total; i += stride) {
    int o = i / K2, j = i - o * K2;
    float v = (j < RANK) ? A[o * RANK + j] : S1[o * NCC + (j - RANK)];
    W2[i] = __float2bfloat16(v);
  }
}

// xg[m,j] = bf16(x[m, col_idx[j]] * wnorm[j]);  H2[m, 512+j2] = bf16(x[m, col_comp_idx[j2]])
__global__ void gather_x(const float* __restrict__ x, const float* __restrict__ wn,
                         const int* __restrict__ cidx, const int* __restrict__ ccidx,
                         __hip_bfloat16* __restrict__ xg, __hip_bfloat16* __restrict__ H2) {
  long long i = (long long)blockIdx.x * blockDim.x + threadIdx.x;
  long long stride = (long long)gridDim.x * blockDim.x;
  const long long total = (long long)MTOT * NFULL;
  for (; i < total; i += stride) {
    int m = (int)(i >> 12);          // NFULL == 4096
    int j = (int)(i & (NFULL - 1));
    if (j < NCK) {
      float v = x[(size_t)m * NFULL + cidx[j]] * wn[j];
      xg[(size_t)m * NCK + j] = __float2bfloat16(v);
    } else {
      int j2 = j - NCK;
      float v = x[(size_t)m * NFULL + ccidx[j2]];
      H2[(size_t)m * K2 + RANK + j2] = __float2bfloat16(v);
    }
  }
}

// ---------------- WMMA GEMM: Out[M,N](+scatter cols) = Act[M,K] @ W[N,K]^T ----------------

union FragU { struct { v8bf lo, hi; } p; v16bf v; };

template<bool A_F32, bool OUT_BF16, bool SCATTER>
__global__ __launch_bounds__(256)
void gemm_wmma_bf16(const void* __restrict__ Aptr, int lda,
                    const __hip_bfloat16* __restrict__ W, int ldw,
                    void* __restrict__ Out, int ldo,
                    const int* __restrict__ colmap, int K) {
  __shared__ __align__(16) __bf16 As[2][BM * LROW];
  __shared__ __align__(16) __bf16 Ws[2][BN * LROW];

  const int tid  = threadIdx.x;
  const int lane = tid & 31;
  const int wav  = tid >> 5;
  const int wm   = wav & 3;       // 4 waves along M, 32 rows each
  const int wnv  = wav >> 2;      // 2 waves along N, 64 cols each
  const int mb   = blockIdx.y * BM;
  const int nb   = blockIdx.x * BN;
  const int l16  = lane & 15;
  const int lh   = lane >> 4;

  v8f   acc[2][4] = {};
  v16bf af[2], wf[4];

  // staging registers (only used by the synchronous paths)
  uint4  wreg[2];
  uint4  areg_h[2];
  float4 areg_f[4];

  // ---- synchronous staging (fallback / f32-convert path) ----
  auto loadW = [&](int k0) {
    #pragma unroll
    for (int i = 0; i < 2; ++i) {
      int id = tid + i * 256, row = id >> 2, seg = id & 3;
      wreg[i] = *(const uint4*)(W + (size_t)(nb + row) * ldw + k0 + seg * 8);
    }
  };
  auto storeW = [&](int buf) {
    #pragma unroll
    for (int i = 0; i < 2; ++i) {
      int id = tid + i * 256, row = id >> 2, seg = id & 3;
      *(uint4*)&Ws[buf][row * LROW + seg * 8] = wreg[i];
    }
  };
  auto loadAh = [&](int k0) {
    const __hip_bfloat16* Ah = (const __hip_bfloat16*)Aptr;
    #pragma unroll
    for (int i = 0; i < 2; ++i) {
      int id = tid + i * 256, row = id >> 2, seg = id & 3;
      areg_h[i] = *(const uint4*)(Ah + (size_t)(mb + row) * lda + k0 + seg * 8);
    }
  };
  auto storeAh = [&](int buf) {
    #pragma unroll
    for (int i = 0; i < 2; ++i) {
      int id = tid + i * 256, row = id >> 2, seg = id & 3;
      *(uint4*)&As[buf][row * LROW + seg * 8] = areg_h[i];
    }
  };
  auto loadAf = [&](int k0) {
    const float* Af = (const float*)Aptr;
    #pragma unroll
    for (int i = 0; i < 4; ++i) {
      int id = tid + i * 256, row = id >> 3, seg = id & 7;
      areg_f[i] = *(const float4*)(Af + (size_t)(mb + row) * lda + k0 + seg * 4);
    }
  };
  auto storeAf = [&](int buf) {
    #pragma unroll
    for (int i = 0; i < 4; ++i) {
      int id = tid + i * 256, row = id >> 3, seg = id & 7;
      uint2 r;
      r.x = (unsigned)f2bf_bits(areg_f[i].x) | ((unsigned)f2bf_bits(areg_f[i].y) << 16);
      r.y = (unsigned)f2bf_bits(areg_f[i].z) | ((unsigned)f2bf_bits(areg_f[i].w) << 16);
      *(uint2*)&As[buf][row * LROW + seg * 4] = r;
    }
  };

  // ---- async staging: global -> LDS with no VGPR round-trip (ASYNCcnt) ----
  auto stageW_async = [&](int buf, int k0) {
    #pragma unroll
    for (int i = 0; i < 2; ++i) {
      int id = tid + i * 256, row = id >> 2, seg = id & 3;
      async_b128(W + (size_t)(nb + row) * ldw + k0 + seg * 8,
                 &Ws[buf][row * LROW + seg * 8]);
    }
  };
  auto stageA_async = [&](int buf, int k0) {
    const __hip_bfloat16* Ah = (const __hip_bfloat16*)Aptr;
    #pragma unroll
    for (int i = 0; i < 2; ++i) {
      int id = tid + i * 256, row = id >> 2, seg = id & 3;
      async_b128(Ah + (size_t)(mb + row) * lda + k0 + seg * 8,
                 &As[buf][row * LROW + seg * 8]);
    }
  };

  // ---- LDS -> WMMA fragments ----
  auto loadFrags = [&](int buf) {
    // A operand (16x32 bf16): lane l16 = row, chunks at K = lh*8 and 16+lh*8
    #pragma unroll
    for (int i = 0; i < 2; ++i) {
      const __bf16* p = &As[buf][(wm * 32 + i * 16 + l16) * LROW + lh * 8];
      FragU f; f.p.lo = *(const v8bf*)p; f.p.hi = *(const v8bf*)(p + 16);
      af[i] = f.v;
    }
    // B operand (32x16 bf16, weights stored [N,K]): lane l16 = out col, chunks at K = lh*8 and 16+lh*8
    #pragma unroll
    for (int j = 0; j < 4; ++j) {
      const __bf16* p = &Ws[buf][(wnv * 64 + j * 16 + l16) * LROW + lh * 8];
      FragU f; f.p.lo = *(const v8bf*)p; f.p.hi = *(const v8bf*)(p + 16);
      wf[j] = f.v;
    }
  };
  auto mma = [&]() {
    #pragma unroll
    for (int i = 0; i < 2; ++i)
      #pragma unroll
      for (int j = 0; j < 4; ++j)
        acc[i][j] = __builtin_amdgcn_wmma_f32_16x16x32_bf16(
            false, af[i], false, wf[j], (short)0, acc[i][j], false, false);
  };

  const int nsteps = K / KS;

  // ---- prologue: stage k-step 0 into buffer 0 ----
#if HAVE_ASYNC_LDS
  if constexpr (A_F32) { loadAf(0); storeAf(0); }
  else                 { stageA_async(0, 0); }
  stageW_async(0, 0);
  wait_async0();
#else
  if constexpr (A_F32) { loadAf(0); storeAf(0); }
  else                 { loadAh(0); storeAh(0); }
  loadW(0); storeW(0);
#endif

  int buf = 0;
  for (int ks = 0; ks < nsteps; ++ks) {
    __syncthreads();
    const bool more = (ks + 1 < nsteps);
    const int  k1   = (ks + 1) * KS;
#if HAVE_ASYNC_LDS
    if (more) {
      if constexpr (A_F32) loadAf(k1);
      else                 stageA_async(buf ^ 1, k1);
      stageW_async(buf ^ 1, k1);
    }
    loadFrags(buf);                       // ds reads of current buffer
    if (more) { if constexpr (A_F32) storeAf(buf ^ 1); }
    mma();                                // 8x v_wmma while async copies land
    if (more) wait_async0();              // our copies into buf^1 done before barrier
#else
    if (more) {
      if constexpr (A_F32) loadAf(k1);
      else                 loadAh(k1);
      loadW(k1);
    }
    loadFrags(buf);                       // frag ds reads first...
    if (more) {                           // ...then retire staging regs before the MMAs
      if constexpr (A_F32) storeAf(buf ^ 1);
      else                 storeAh(buf ^ 1);
      storeW(buf ^ 1);
    }
    mma();
#endif
    buf ^= 1;
  }

  // ---- epilogue: D layout — lane l16 = col, VGPR v = row (v + 8*lh) within 16x16 tile ----
  #pragma unroll
  for (int j = 0; j < 4; ++j) {
    int c  = nb + wnv * 64 + j * 16 + l16;
    int oc = SCATTER ? colmap[c] : c;
    #pragma unroll
    for (int i = 0; i < 2; ++i) {
      int r0 = mb + wm * 32 + i * 16 + lh * 8;
      #pragma unroll
      for (int v = 0; v < 8; ++v) {
        size_t off = (size_t)(r0 + v) * ldo + oc;
        if constexpr (OUT_BF16)
          ((__hip_bfloat16*)Out)[off] = __float2bfloat16(acc[i][j][v]);
        else
          ((float*)Out)[off] = acc[i][j][v];
      }
    }
  }
}

// ---------------- launch ----------------

extern "C" void kernel_launch(void* const* d_in, const int* in_sizes, int n_in,
                              void* d_out, int out_size, void* d_ws, size_t ws_size,
                              hipStream_t stream) {
  (void)in_sizes; (void)n_in; (void)out_size; (void)ws_size;
  const float* x    = (const float*)d_in[0];
  const float* A    = (const float*)d_in[1];
  const float* B    = (const float*)d_in[2];
  const float* S1   = (const float*)d_in[3];
  const float* S2   = (const float*)d_in[4];
  const float* wn   = (const float*)d_in[5];
  const int* cidx   = (const int*)d_in[6];
  const int* ccidx  = (const int*)d_in[7];
  const int* ridx   = (const int*)d_in[8];
  const int* rcidx  = (const int*)d_in[9];
  float* out = (float*)d_out;

  // workspace layout (bf16), ~84 MB total
  char* ws = (char*)d_ws;
  size_t off = 0;
  __hip_bfloat16* xg  = (__hip_bfloat16*)(ws + off); off += (size_t)MTOT * NCK * 2;
  __hip_bfloat16* H2  = (__hip_bfloat16*)(ws + off); off += (size_t)MTOT * K2  * 2;
  __hip_bfloat16* Bw  = (__hip_bfloat16*)(ws + off); off += (size_t)RANK * NCK * 2;
  __hip_bfloat16* W2  = (__hip_bfloat16*)(ws + off); off += (size_t)NCK  * K2  * 2;
  __hip_bfloat16* S2b = (__hip_bfloat16*)(ws + off);

  cvt_f32_bf16<<<2048, 256, 0, stream>>>(B,  Bw,  RANK * NCK);
  cvt_f32_bf16<<<2048, 256, 0, stream>>>(S2, S2b, NCC * NFULL);
  build_w2   <<<4096, 256, 0, stream>>>(A, S1, W2);
  gather_x   <<<8192, 256, 0, stream>>>(x, wn, cidx, ccidx, xg, H2);

  dim3 blk(256);
  // G1: hidden(H2[:,0:512]) = xg @ Bw^T           [8192x512, K=3840], bf16 out
  gemm_wmma_bf16<false, true,  false><<<dim3(RANK / BN, MTOT / BM), blk, 0, stream>>>(
      xg, NCK, Bw, NCK, H2, K2, nullptr, NCK);
  // G2: y[:, row_idx] = H2 @ [A|S1]^T             [8192x3840, K=768], f32 scatter
  gemm_wmma_bf16<false, false, true ><<<dim3(NCK / BN, MTOT / BM), blk, 0, stream>>>(
      H2, K2, W2, K2, out, NFULL, ridx, K2);
  // G3: y[:, row_comp_idx] = x @ S2^T             [8192x256, K=4096], f32 scatter, fp32 A cvt in staging
  gemm_wmma_bf16<true,  false, true ><<<dim3(NCC / BN, MTOT / BM), blk, 0, stream>>>(
      x, NFULL, S2b, NFULL, out, NFULL, rcidx, NFULL);
}
